// SelfAttention_65816078844258
// MI455X (gfx1250) — compile-verified, matrix operand order checked by reading, not run on hardware
//
#include <hip/hip_runtime.h>

// ---------------------------------------------------------------------------
// Types for CDNA5 WMMA (wave32): bf16 16x16x32, fp32 accumulate.
// ---------------------------------------------------------------------------
typedef __attribute__((ext_vector_type(16))) __bf16 v16bf;
typedef __attribute__((ext_vector_type(8)))  float  v8f;

struct __attribute__((aligned(16))) u32x4 { unsigned x, y, z, w; };

union Frag {
  v16bf v;
  u32x4 q[2];
  unsigned short s[16];
};

#define WMMA_BF16(A, B, C) \
  __builtin_amdgcn_wmma_f32_16x16x32_bf16(false, (A), false, (B), (short)0, (C), false, false)

__device__ __forceinline__ unsigned short f2bf(float f) {
  union { float f; unsigned u; } c; c.f = f;
  unsigned u = c.u;
  return (unsigned short)((u + 0x7FFFu + ((u >> 16) & 1u)) >> 16);
}

// ---------------------------------------------------------------------------
// fp32 -> bf16 bulk convert
// ---------------------------------------------------------------------------
__global__ void cvt_f32_bf16(const float* __restrict__ in,
                             unsigned short* __restrict__ out, int n) {
  int i = blockIdx.x * blockDim.x + threadIdx.x;
  if (i < n) out[i] = f2bf(in[i]);
}

// ---------------------------------------------------------------------------
// T5 relative-position bias table: biastab[h][rel + 2047], rel in [-2047,2047]
// ---------------------------------------------------------------------------
__global__ void bias_table_kernel(const float* __restrict__ rel_bias_table, // [32,16]
                                  float* __restrict__ biastab) {            // [16,4095]
  int i = blockIdx.x * blockDim.x + threadIdx.x;
  if (i >= 16 * 4095) return;
  int h   = i / 4095;
  int rel = (i % 4095) - 2047;              // rel = key - query
  int bucket = (rel > 0) ? 16 : 0;
  int rp = rel < 0 ? -rel : rel;
  int v;
  if (rp < 8) {
    v = rp;
  } else {
    v = 8 + (int)(logf((float)rp * 0.125f) * (8.0f / logf(16.0f)));
    if (v > 15) v = 15;
  }
  biastab[i] = rel_bias_table[(bucket + v) * 16 + h];
}

// ---------------------------------------------------------------------------
// V transpose: qkv[b,n,2,h,d] (bf16) -> vt[b,h,d,n] (bf16) so P@V B-fragments
// load contiguously along keys.
// ---------------------------------------------------------------------------
__global__ void transpose_v(const unsigned short* __restrict__ qkv,
                            unsigned short* __restrict__ vt) {
  int i = blockIdx.x * blockDim.x + threadIdx.x;   // over 4*16*64*2048
  int n = i & 2047;
  int d = (i >> 11) & 63;
  int h = (i >> 17) & 15;
  int b = i >> 21;
  vt[i] = qkv[(size_t)(b * 2048 + n) * 3072 + 2048 + h * 64 + d];
}

// ---------------------------------------------------------------------------
// Register-tiled bf16 GEMM: C[M,Nc] = A[M,K] @ W[Nc,K]^T  (+bias).
// Block = 128 threads (4 waves) covering a 64x128 block tile; each wave owns a
// 32x64 tile (2x4 wmma accums -> 8 wmma per K-step of 32, 12 b128 loads).
// Grid: (M/64, Nc/128).  outb!=null -> bf16 output; else f32 output (+bias).
// ---------------------------------------------------------------------------
__global__ void gemm_bf16(const unsigned short* __restrict__ A,
                          const unsigned short* __restrict__ W,
                          unsigned short* __restrict__ outb,
                          float* __restrict__ outf,
                          const float* __restrict__ bias,
                          int K, int Nc) {
  const int lane = threadIdx.x & 31;
  const int wave = threadIdx.x >> 5;
  const int rowBase = blockIdx.x * 64 + (wave >> 1) * 32;
  const int colBase = blockIdx.y * 128 + (wave & 1) * 64;
  const int rl = lane & 15;
  const int kk = (lane & 16) ? 8 : 0;   // per-lane K sub-offset (ISA 16-bit A/B layout)

  const u32x4* arow0 = (const u32x4*)(A + (size_t)(rowBase + rl)      * K);
  const u32x4* arow1 = (const u32x4*)(A + (size_t)(rowBase + 16 + rl) * K);
  const u32x4* wrow[4];
#pragma unroll
  for (int t = 0; t < 4; ++t)
    wrow[t] = (const u32x4*)(W + (size_t)(colBase + t * 16 + rl) * K);

  v8f c[2][4];
  const v8f vzero = {0.f, 0.f, 0.f, 0.f, 0.f, 0.f, 0.f, 0.f};
#pragma unroll
  for (int i = 0; i < 2; ++i)
#pragma unroll
    for (int t = 0; t < 4; ++t) c[i][t] = vzero;

  for (int k0 = 0; k0 < K; k0 += 32) {
    int bi = (k0 + kk) >> 3;            // index in 8-element (16B) units
    __builtin_prefetch(arow0 + bi + 8, 0, 0);
    Frag a0, a1;
    a0.q[0] = arow0[bi]; a0.q[1] = arow0[bi + 2];
    a1.q[0] = arow1[bi]; a1.q[1] = arow1[bi + 2];
#pragma unroll
    for (int t = 0; t < 4; ++t) {
      Frag bt;
      bt.q[0] = wrow[t][bi]; bt.q[1] = wrow[t][bi + 2];
      c[0][t] = WMMA_BF16(a0.v, bt.v, c[0][t]);
      c[1][t] = WMMA_BF16(a1.v, bt.v, c[1][t]);
    }
  }

  const int rsub = (lane & 16) ? 8 : 0; // C/D layout: row = j + rsub, col = rl
  if (outb) {
#pragma unroll
    for (int i = 0; i < 2; ++i)
#pragma unroll
      for (int t = 0; t < 4; ++t)
#pragma unroll
        for (int j = 0; j < 8; ++j) {
          int r = rowBase + i * 16 + rsub + j;
          int cc = colBase + t * 16 + rl;
          outb[(size_t)r * Nc + cc] = f2bf(c[i][t][j]);
        }
  } else {
#pragma unroll
    for (int i = 0; i < 2; ++i)
#pragma unroll
      for (int t = 0; t < 4; ++t)
#pragma unroll
        for (int j = 0; j < 8; ++j) {
          int r = rowBase + i * 16 + rsub + j;
          int cc = colBase + t * 16 + rl;
          outf[(size_t)r * Nc + cc] = c[i][t][j] + bias[cc];
        }
  }
}

// ---------------------------------------------------------------------------
// Flash attention: grid (N/128, H, B), block 128 (4 waves), each wave owns 32
// query rows (2 row tiles). Keys in blocks of 32: K/V fragments loaded once and
// reused across both row tiles -> 16 wmma per key block. Online softmax via
// lane butterflies; bias row staged in LDS; P staged through per-wave LDS as
// A-fragments. Output bf16 at [b, n, h*64+d].
// ---------------------------------------------------------------------------
__global__ void attn_kernel(const unsigned short* __restrict__ qkv, // [B,N,3,H,D] bf16
                            const unsigned short* __restrict__ vt,  // [B,H,D,N] bf16
                            const float* __restrict__ biastab,      // [H, 4095]
                            unsigned short* __restrict__ attnout) { // [B,N,E] bf16
  constexpr int N = 2048, H = 16;
  __shared__ __align__(16) unsigned short psh[4][2][16 * 32];
  __shared__ float bsh[4095];

  const int lane  = threadIdx.x & 31;
  const int wave  = threadIdx.x >> 5;
  const int qbase = blockIdx.x * 128 + wave * 32;
  const int h = blockIdx.y, b = blockIdx.z;
  const int rl = lane & 15;
  const int kk = (lane & 16) ? 8 : 0;
  const int rsub = (lane & 16) ? 8 : 0;
  const int bi = kk >> 3;

  const unsigned short* Qp = qkv + (size_t)b * N * 3072 + h * 64;          // + r*3072
  const unsigned short* Kp = qkv + (size_t)b * N * 3072 + 1024 + h * 64;   // + key*3072
  const unsigned short* Vt = vt + (size_t)(b * H + h) * 64 * N;            // + d*N + key

  // stage this head's bias row (16KB) into LDS
  for (int i = threadIdx.x; i < 4095; i += 128) bsh[i] = biastab[h * 4095 + i];
  __syncthreads();

  // Q A-fragments: [row tile r][d-half]; resident for the whole key loop.
  Frag aq[2][2];
#pragma unroll
  for (int r = 0; r < 2; ++r) {
    const u32x4* qr = (const u32x4*)(Qp + (size_t)(qbase + r * 16 + rl) * 3072);
    aq[r][0].q[0] = qr[bi];     aq[r][0].q[1] = qr[bi + 2];
    aq[r][1].q[0] = qr[bi + 4]; aq[r][1].q[1] = qr[bi + 6];
  }

  float m[2][8], l[2][8];
  v8f o[2][4];
  const v8f vzero = {0.f, 0.f, 0.f, 0.f, 0.f, 0.f, 0.f, 0.f};
#pragma unroll
  for (int r = 0; r < 2; ++r) {
#pragma unroll
    for (int j = 0; j < 8; ++j) { m[r][j] = -1e30f; l[r][j] = 0.0f; }
#pragma unroll
    for (int t = 0; t < 4; ++t) o[r][t] = vzero;
  }

  const float scale = 0.125f;   // D^-0.5, D=64

  for (int kb = 0; kb < N; kb += 32) {
    // ---- S = Q @ K^T : 2 row tiles x 2 col tiles of 16x16 scores ----
    v8f s[2][2];
#pragma unroll
    for (int r = 0; r < 2; ++r) { s[r][0] = vzero; s[r][1] = vzero; }
    {
      const u32x4* kr0 = (const u32x4*)(Kp + (size_t)(kb + rl) * 3072);
      const u32x4* kr1 = (const u32x4*)(Kp + (size_t)(kb + 16 + rl) * 3072);
      Frag bk0, bk1;
      bk0.q[0] = kr0[bi]; bk0.q[1] = kr0[bi + 2];       // d = 0..31
      bk1.q[0] = kr1[bi]; bk1.q[1] = kr1[bi + 2];
#pragma unroll
      for (int r = 0; r < 2; ++r) {
        s[r][0] = WMMA_BF16(aq[r][0].v, bk0.v, s[r][0]);
        s[r][1] = WMMA_BF16(aq[r][0].v, bk1.v, s[r][1]);
      }
      bk0.q[0] = kr0[bi + 4]; bk0.q[1] = kr0[bi + 6];   // d = 32..63
      bk1.q[0] = kr1[bi + 4]; bk1.q[1] = kr1[bi + 6];
#pragma unroll
      for (int r = 0; r < 2; ++r) {
        s[r][0] = WMMA_BF16(aq[r][1].v, bk0.v, s[r][0]);
        s[r][1] = WMMA_BF16(aq[r][1].v, bk1.v, s[r][1]);
      }
    }

    // ---- bias + scale, online softmax update ----
#pragma unroll
    for (int r = 0; r < 2; ++r) {
      unsigned short* pw = &psh[wave][r][0];
#pragma unroll
      for (int j = 0; j < 8; ++j) {
        int row  = qbase + r * 16 + rsub + j;
        int key0 = kb + rl;
        float f0 = s[r][0][j] * scale + bsh[key0 - row + 2047];
        float f1 = s[r][1][j] * scale + bsh[key0 + 16 - row + 2047];
        float rmax = fmaxf(f0, f1);
        rmax = fmaxf(rmax, __shfl_xor(rmax, 1, 32));
        rmax = fmaxf(rmax, __shfl_xor(rmax, 2, 32));
        rmax = fmaxf(rmax, __shfl_xor(rmax, 4, 32));
        rmax = fmaxf(rmax, __shfl_xor(rmax, 8, 32));
        float mnew  = fmaxf(m[r][j], rmax);
        float alpha = __expf(m[r][j] - mnew);
        float p0 = __expf(f0 - mnew);
        float p1 = __expf(f1 - mnew);
        float rsum = p0 + p1;
        rsum += __shfl_xor(rsum, 1, 32);
        rsum += __shfl_xor(rsum, 2, 32);
        rsum += __shfl_xor(rsum, 4, 32);
        rsum += __shfl_xor(rsum, 8, 32);
        l[r][j] = l[r][j] * alpha + rsum;
        m[r][j] = mnew;
        o[r][0][j] *= alpha; o[r][1][j] *= alpha;
        o[r][2][j] *= alpha; o[r][3][j] *= alpha;
        pw[(rsub + j) * 32 + rl]      = f2bf(p0);
        pw[(rsub + j) * 32 + rl + 16] = f2bf(p1);
      }
    }
    __syncthreads();   // order P stores before fragment reload

    // ---- P (16x32 per row tile) as A-fragments from LDS ----
    Frag ap[2];
#pragma unroll
    for (int r = 0; r < 2; ++r) {
      const u32x4* pr = (const u32x4*)(&psh[wave][r][rl * 32]);
      ap[r].q[0] = pr[bi]; ap[r].q[1] = pr[bi + 2];
    }
    // ---- O += P @ V : 4 D-tiles of 16, V fragment reused across row tiles ----
#pragma unroll
    for (int t = 0; t < 4; ++t) {
      const u32x4* vr = (const u32x4*)(Vt + (size_t)(t * 16 + rl) * N + kb);
      Frag bv;
      bv.q[0] = vr[bi]; bv.q[1] = vr[bi + 2];
      o[0][t] = WMMA_BF16(ap[0].v, bv.v, o[0][t]);
      o[1][t] = WMMA_BF16(ap[1].v, bv.v, o[1][t]);
    }
    __syncthreads();   // protect LDS P tiles before next iteration's stores
  }

  // ---- normalize and store bf16 at [b, row, h*64 + d] ----
#pragma unroll
  for (int r = 0; r < 2; ++r)
#pragma unroll
    for (int j = 0; j < 8; ++j) {
      float inv = 1.0f / l[r][j];
      int row = qbase + r * 16 + rsub + j;
#pragma unroll
      for (int t = 0; t < 4; ++t) {
        int d = h * 64 + t * 16 + rl;
        attnout[(size_t)(b * N + row) * 1024 + d] = f2bf(o[r][t][j] * inv);
      }
    }
}

// ---------------------------------------------------------------------------
// Launch: cvt x/wqkv/wproj -> bias table -> QKV GEMM -> V transpose ->
//         flash attention -> proj GEMM (+bias, fp32 out).
// ---------------------------------------------------------------------------
extern "C" void kernel_launch(void* const* d_in, const int* in_sizes, int n_in,
                              void* d_out, int out_size, void* d_ws, size_t ws_size,
                              hipStream_t stream) {
  const float* x        = (const float*)d_in[0];   // [4,2048,1024]
  const float* w_qkv    = (const float*)d_in[1];   // [3072,1024]
  const float* w_proj   = (const float*)d_in[2];   // [1024,1024]
  const float* b_proj   = (const float*)d_in[3];   // [1024]
  const float* rel_tab  = (const float*)d_in[4];   // [32,16]
  float* out = (float*)d_out;                      // [4,2048,1024]

  char* ws = (char*)d_ws;
  unsigned short* xbf     = (unsigned short*)(ws + 0);           // 16,777,216 B
  unsigned short* wqkvbf  = (unsigned short*)(ws + 16777216);    //  6,291,456 B
  unsigned short* wprojbf = (unsigned short*)(ws + 23068672);    //  2,097,152 B
  unsigned short* qkvbf   = (unsigned short*)(ws + 25165824);    // 50,331,648 B
  unsigned short* vtbf    = (unsigned short*)(ws + 75497472);    // 16,777,216 B
  unsigned short* attnbf  = (unsigned short*)(ws + 92274688);    // 16,777,216 B
  float*          biastab = (float*)(ws + 109051904);            //    262,080 B

  cvt_f32_bf16<<<(8388608 + 255) / 256, 256, 0, stream>>>(x, xbf, 8388608);
  cvt_f32_bf16<<<(3145728 + 255) / 256, 256, 0, stream>>>(w_qkv, wqkvbf, 3145728);
  cvt_f32_bf16<<<(1048576 + 255) / 256, 256, 0, stream>>>(w_proj, wprojbf, 1048576);
  bias_table_kernel<<<(16 * 4095 + 255) / 256, 256, 0, stream>>>(rel_tab, biastab);

  // qkv[8192,3072] = x[8192,1024] @ w_qkv^T  (bf16 out)
  gemm_bf16<<<dim3(8192 / 64, 3072 / 128), 128, 0, stream>>>(
      xbf, wqkvbf, qkvbf, nullptr, nullptr, 1024, 3072);

  transpose_v<<<8388608 / 256, 256, 0, stream>>>(qkvbf, vtbf);

  attn_kernel<<<dim3(2048 / 128, 16, 4), 128, 0, stream>>>(
      qkvbf, vtbf, biastab, attnbf);

  // out[8192,1024] = attn[8192,1024] @ w_proj^T + b_proj  (fp32 out)
  gemm_bf16<<<dim3(8192 / 64, 1024 / 128), 128, 0, stream>>>(
      attnbf, wprojbf, nullptr, out, b_proj, 1024, 1024);
}
